// DDGCNet1_23089744183607
// MI455X (gfx1250) — matrix operand order repaired and checked by reading, not compiled, not run on hardware
//
#include <hip/hip_runtime.h>

#define NPTS 4096
#define BB 2
#define PTOT (BB * NPTS) // 8192

typedef _Float16 __attribute__((ext_vector_type(16))) v16h;
typedef _Float16 __attribute__((ext_vector_type(8))) v8h;
typedef _Float16 __attribute__((ext_vector_type(4))) v4h;
typedef float __attribute__((ext_vector_type(8))) v8f;

static inline unsigned cdiv(long a, long b) { return (unsigned)((a + b - 1) / b); }
static inline int rup(int x, int m) { return (x + m - 1) / m * m; }

// ---------------- elementwise utils ----------------
__global__ void k_zero(float* p, int n) {
  int t = blockIdx.x * 256 + threadIdx.x;
  if (t < n) p[t] = 0.f;
}
__global__ void k_mul(float* a, const float* __restrict__ b, long n) {
  long t = (long)blockIdx.x * 256 + threadIdx.x;
  if (t < n) a[t] *= b[t];
}
__global__ void k_add(float* a, const float* __restrict__ b, long n) {
  long t = (long)blockIdx.x * 256 + threadIdx.x;
  if (t < n) a[t] += b[t];
}

// -------- weight pack: f32 [K x N] -> f16 n-major [Npad x Kpad], zero padded --------
__global__ void k_pack(const float* __restrict__ W, _Float16* __restrict__ Wp,
                       int K, int N, int Kpad, int Npad) {
  int t = blockIdx.x * 256 + threadIdx.x;
  if (t >= Kpad * Npad) return;
  int n = t / Kpad, k = t % Kpad;
  float v = (k < K && n < N) ? W[k * N + n] : 0.f;
  Wp[t] = (_Float16)v;
}

// ---------------- WMMA GEMM: Out[M x Nc] = A[M x K] * W[K x N] (+bias) ----------------
// Block tile 128(M) x 64(N); 8 waves = 4(M) x 2(N); each wave owns a 2x2 grid of
// 16x16 WMMA tiles (32x32 output) => each fragment feeds 2 WMMAs. K-tile 64.
// MODE 0: A dense f32, leading dim lda.
// MODE 2: edge features: row r -> (p=r/Knb, kk=r%Knb);
//         A[r,c] = c<C ? X[p*ldx+c] : X[j*ldx+(c-C)] - X[p*ldx+(c-C)], j=eidx[p*Knb+kk]
template <int MODE>
__global__ __launch_bounds__(256) void k_gemm(
    const float* __restrict__ A, int lda,
    const int* __restrict__ eidx, int Knb, int C, int ldx,
    const _Float16* __restrict__ WpT, const float* __restrict__ bias,
    float* __restrict__ Out, int M, int Kpad, int Kact, int Nc, int Npad,
    int ldo, int ocol) {
  __shared__ _Float16 As[128 * 64];  // [row][k], row stride 64 halves (128B)
  __shared__ _Float16 WsT[64 * 64];  // [n][k]
  __shared__ int ra[128];
  __shared__ int rb[128];
  int tid = threadIdx.x;
  int m0 = blockIdx.x * 128;
  int n0 = blockIdx.y * 64;

  if (MODE == 2) {
    if (tid < 128) {
      int grow = m0 + tid;
      if (grow >= M) grow = M - 1; // clamp; OOB rows masked at store
      int p = grow / Knb;
      ra[tid] = p * ldx;
      rb[tid] = eidx[p * Knb + (grow % Knb)] * ldx;
    }
    __syncthreads();
  }

  int wid = tid >> 5, lane = tid & 31;
  int wm = wid & 3, wn = wid >> 2;
  int hi = lane >> 4, lm = lane & 15;

  v8f acc00 = {0.f, 0.f, 0.f, 0.f, 0.f, 0.f, 0.f, 0.f};
  v8f acc01 = acc00, acc10 = acc00, acc11 = acc00;
  bool vec_ok = (MODE == 0) && ((lda & 3) == 0);

  for (int k0 = 0; k0 < Kpad; k0 += 64) {
    // ---- stage A tile 128x64 (f32 -> f16) ----
    if (MODE == 0) {
      for (int r = 0; r < 8; ++r) {
        int linear = (r * 256 + tid) * 4;
        int row = linear >> 6, kk = linear & 63;
        int grow = m0 + row, gk = k0 + kk;
        float4 v = {0.f, 0.f, 0.f, 0.f};
        if (grow < M) {
          const float* ap = A + (long)grow * lda + gk;
          if (vec_ok && gk + 3 < Kact) {
            v = *(const float4*)ap;
          } else {
            if (gk + 0 < Kact) v.x = ap[0];
            if (gk + 1 < Kact) v.y = ap[1];
            if (gk + 2 < Kact) v.z = ap[2];
            if (gk + 3 < Kact) v.w = ap[3];
          }
          // pull next K-tile of this row toward the caches (global_prefetch_b8)
          if (gk + 64 < Kact) __builtin_prefetch(ap + 64, 0, 1);
        }
        v4h h4 = {(_Float16)v.x, (_Float16)v.y, (_Float16)v.z, (_Float16)v.w};
        *(v4h*)&As[row * 64 + kk] = h4;
      }
    } else {
      for (int r = 0; r < 32; ++r) {
        int linear = r * 256 + tid;
        int row = linear >> 6, kk = linear & 63;
        int gk = k0 + kk;
        int cm = gk < C ? gk : gk - C;
        if (cm >= C) cm = 0; // pad region, value forced to 0 below
        float vi = A[ra[row] + cm];
        float vj = A[rb[row] + cm];
        float v = gk < C ? vi : (gk < 2 * C ? vj - vi : 0.f);
        As[row * 64 + kk] = (_Float16)v;
      }
    }
    // ---- stage W tile 64(N) x 64(K), K-contiguous per column ----
    for (int r = 0; r < 4; ++r) {
      int linear = (r * 256 + tid) * 4;
      int nn = linear >> 6, kk = linear & 63;
      int gn = n0 + nn;
      v4h w = {(_Float16)0.f, (_Float16)0.f, (_Float16)0.f, (_Float16)0.f};
      if (gn < Npad) {
        const _Float16* wp = WpT + (long)gn * Kpad + k0 + kk;
        w = *(const v4h*)wp;
        if (k0 + 64 < Kpad) __builtin_prefetch(wp + 64, 0, 1);
      }
      *(v4h*)&WsT[nn * 64 + kk] = w;
    }
    __syncthreads();

    // ---- fragments: contiguous 16B runs -> ds_load_b128; 2x2 register blocking ----
    const v8h* Ar0 = (const v8h*)&As[(wm * 32 + lm) * 64];
    const v8h* Ar1 = (const v8h*)&As[(wm * 32 + 16 + lm) * 64];
    const v8h* Br0 = (const v8h*)&WsT[(wn * 32 + lm) * 64];
    const v8h* Br1 = (const v8h*)&WsT[(wn * 32 + 16 + lm) * 64];
#pragma unroll
    for (int ks = 0; ks < 2; ++ks) { // two 32-wide K sub-tiles
      v16h a0 = __builtin_shufflevector(Ar0[ks * 4 + hi], Ar0[ks * 4 + 2 + hi],
                                        0, 1, 2, 3, 4, 5, 6, 7, 8, 9, 10, 11, 12, 13, 14, 15);
      v16h a1 = __builtin_shufflevector(Ar1[ks * 4 + hi], Ar1[ks * 4 + 2 + hi],
                                        0, 1, 2, 3, 4, 5, 6, 7, 8, 9, 10, 11, 12, 13, 14, 15);
      v16h b0 = __builtin_shufflevector(Br0[ks * 4 + hi], Br0[ks * 4 + 2 + hi],
                                        0, 1, 2, 3, 4, 5, 6, 7, 8, 9, 10, 11, 12, 13, 14, 15);
      v16h b1 = __builtin_shufflevector(Br1[ks * 4 + hi], Br1[ks * 4 + 2 + hi],
                                        0, 1, 2, 3, 4, 5, 6, 7, 8, 9, 10, 11, 12, 13, 14, 15);
      acc00 = __builtin_amdgcn_wmma_f32_16x16x32_f16(false, a0, false, b0, (short)0, acc00, false, false);
      acc01 = __builtin_amdgcn_wmma_f32_16x16x32_f16(false, a0, false, b1, (short)0, acc01, false, false);
      acc10 = __builtin_amdgcn_wmma_f32_16x16x32_f16(false, a1, false, b0, (short)0, acc10, false, false);
      acc11 = __builtin_amdgcn_wmma_f32_16x16x32_f16(false, a1, false, b1, (short)0, acc11, false, false);
    }
    __syncthreads();
  }

#pragma unroll
  for (int ti = 0; ti < 2; ++ti) {
#pragma unroll
    for (int tj = 0; tj < 2; ++tj) {
      v8f acc = ti == 0 ? (tj == 0 ? acc00 : acc01) : (tj == 0 ? acc10 : acc11);
      int colb = n0 + wn * 32 + tj * 16 + lm;
#pragma unroll
      for (int r = 0; r < 8; ++r) {
        int mrow = m0 + wm * 32 + ti * 16 + hi * 8 + r; // C/D f32 layout
        if (mrow < M && colb < Nc) {
          float o = acc[r] + (bias ? bias[colb] : 0.f);
          Out[(long)mrow * ldo + ocol + colb] = o;
        }
      }
    }
  }
}

// ------- per-row distance + full bitonic sort, emit first maxs neighbor indices -------
__global__ __launch_bounds__(256) void k_sortknn(const float* __restrict__ X, int ldx, int C,
                                                 int* __restrict__ outS, int ostride, int maxs,
                                                 int Nn) {
  __shared__ float key[NPTS];
  __shared__ int vid[NPTS];
  __shared__ float xi[256];
  int p = blockIdx.x;
  int b = p / Nn;
  long base = (long)p * ldx;
  for (int c = threadIdx.x; c < C; c += 256) xi[c] = X[base + c];
  __syncthreads();
  for (int j = threadIdx.x; j < Nn; j += 256) {
    const float* xj = X + (long)(b * Nn + j) * ldx;
    float d = 0.f;
    for (int c = 0; c < C; ++c) {
      float t = xi[c] - xj[c];
      d += t * t;
    }
    key[j] = d;
    vid[j] = j;
  }
  __syncthreads();
  for (int k = 2; k <= Nn; k <<= 1) {
    for (int j = k >> 1; j > 0; j >>= 1) {
      for (int i = threadIdx.x; i < Nn; i += 256) {
        int ixj = i ^ j;
        if (ixj > i) {
          bool up = ((i & k) == 0);
          float ki = key[i], kx = key[ixj];
          if (up ? (ki > kx) : (ki < kx)) {
            key[i] = kx;
            key[ixj] = ki;
            int t = vid[i];
            vid[i] = vid[ixj];
            vid[ixj] = t;
          }
        }
      }
      __syncthreads();
    }
  }
  for (int t = threadIdx.x; t < maxs; t += 256)
    outS[(long)p * ostride + t] = b * Nn + vid[t]; // global row index
}

// ------- assemble concatenated neighbor lists from sorted tables -------
// seg.x: 0=feature table, 1=pos table, -1=unused; seg.y=count; seg.z=stride
__global__ void k_bldidx(const int* __restrict__ fdS, const int* __restrict__ cdS, int sstride,
                         int* __restrict__ out, int outK, int P,
                         int4 s0, int4 s1, int4 s2, int4 s3) {
  int t = blockIdx.x * 256 + threadIdx.x;
  if (t >= P * outK) return;
  int p = t / outK, q = t % outK;
  int4 segs[4] = {s0, s1, s2, s3};
  int off = 0, val = 0;
  for (int s = 0; s < 4; ++s) {
    if (segs[s].x < 0) continue;
    int cnt = segs[s].y;
    if (q < off + cnt) {
      const int* tab = segs[s].x ? cdS : fdS;
      val = tab[(long)p * sstride + (q - off) * segs[s].z];
      break;
    }
    off += cnt;
  }
  out[t] = val;
}

// ------- batch-norm statistics over M rows (training mode) -------
__global__ __launch_bounds__(256) void k_bnstats(const float* __restrict__ X, long M, int Nc,
                                                 float* __restrict__ st) {
  __shared__ float s1[1024];
  __shared__ float s2[1024];
  for (int c = threadIdx.x; c < Nc; c += 256) { s1[c] = 0.f; s2[c] = 0.f; }
  __syncthreads();
  for (long r = blockIdx.x; r < M; r += gridDim.x) {
    const float* row = X + r * Nc;
    for (int c = threadIdx.x; c < Nc; c += 256) {
      float v = row[c];
      s1[c] += v;
      s2[c] += v * v;
    }
  }
  __syncthreads();
  for (int c = threadIdx.x; c < Nc; c += 256) {
    atomicAdd(&st[c], s1[c]);
    atomicAdd(&st[Nc + c], s2[c]);
  }
}

// act: 0=none 1=lrelu(0.2) 2=relu 3=sigmoid
__global__ void k_bnact(float* X, long M, int Nc, const float* __restrict__ st,
                        const float* __restrict__ g, const float* __restrict__ be,
                        float invM, int act) {
  long t = (long)blockIdx.x * 256 + threadIdx.x;
  if (t >= M * Nc) return;
  int c = (int)(t % Nc);
  float mean = st[c] * invM;
  float var = st[Nc + c] * invM - mean * mean;
  float v = X[t];
  v = g[c] * (v - mean) * rsqrtf(var + 1e-5f) + be[c];
  if (act == 1) v = v > 0.f ? v : 0.2f * v;
  else if (act == 2) v = fmaxf(v, 0.f);
  else if (act == 3) v = 1.f / (1.f + expf(-v));
  X[t] = v;
}

// fused BN + lrelu + max over K neighbors, scatter into xcat column block
__global__ void k_bnmaxk(const float* __restrict__ E, int P, int Knb, int Nc,
                         const float* __restrict__ st, const float* __restrict__ g,
                         const float* __restrict__ be, float invM,
                         float* __restrict__ Out, int ldo, int ocol) {
  int t = blockIdx.x * 256 + threadIdx.x;
  if (t >= P * Nc) return;
  int p = t / Nc, c = t % Nc;
  float mean = st[c] * invM;
  float is = rsqrtf(st[Nc + c] * invM - mean * mean + 1e-5f);
  float gg = g[c], bb = be[c];
  float m = -3.4e38f;
  const float* base = E + (long)p * Knb * Nc + c;
  for (int k = 0; k < Knb; ++k) {
    float v = gg * (base[(long)k * Nc] - mean) * is + bb;
    v = v > 0.f ? v : 0.2f * v;
    m = fmaxf(m, v);
  }
  Out[(long)p * ldo + ocol + c] = m;
}

__global__ void k_maxpool(const float* __restrict__ H, int Bv, int Nn, int C, float* __restrict__ G) {
  int t = blockIdx.x * 256 + threadIdx.x;
  if (t >= Bv * C) return;
  int b = t / C, c = t % C;
  float m = -3.4e38f;
  const float* base = H + (long)b * Nn * C + c;
  for (int n = 0; n < Nn; ++n) m = fmaxf(m, base[(long)n * C]);
  G[t] = m;
}

__global__ void k_stnapply(const float* __restrict__ X, const float* __restrict__ T,
                           float* __restrict__ Y, int P, int Nn) {
  int t = blockIdx.x * 256 + threadIdx.x;
  if (t >= P * 15) return;
  int p = t / 15, d = t % 15;
  const float* tb = T + (p / Nn) * 225;
  float s = 0.f;
  for (int c = 0; c < 15; ++c)
    s += X[p * 15 + c] * (tb[c * 15 + d] + (c == d ? 1.f : 0.f));
  Y[t] = s;
}

// =====================================================================================
extern "C" void kernel_launch(void* const* d_in, const int* in_sizes, int n_in,
                              void* d_out, int out_size, void* d_ws, size_t ws_size,
                              hipStream_t stream) {
  (void)in_sizes; (void)n_in; (void)out_size; (void)ws_size;
  const float* x = (const float*)d_in[0];
  const float* pos = (const float*)d_in[1];
  auto F = [&](int i) -> const float* { return (const float*)d_in[i]; };

  // ---- carve workspace ----
  char* wsc = (char*)d_ws;
  auto alloc = [&](size_t bytes) -> void* {
    void* p = wsc;
    wsc += (bytes + 255) & ~(size_t)255;
    return p;
  };
  int* cdS = (int*)alloc((size_t)PTOT * 544 * 4);
  int* fdS = (int*)alloc((size_t)PTOT * 544 * 4);
  int* idxb = (int*)alloc((size_t)PTOT * 44 * 4);
  int* idxf = (int*)alloc((size_t)PTOT * 44 * 4);
  float* xstn = (float*)alloc((size_t)PTOT * 15 * 4);
  float* xcat = (float*)alloc((size_t)PTOT * 380 * 4);
  float* E1 = (float*)alloc((size_t)PTOT * 44 * 120 * 4);
  float* E2 = (float*)alloc((size_t)PTOT * 44 * 120 * 4);
  float* h64 = (float*)alloc((size_t)PTOT * 64 * 4);
  float* h128 = (float*)alloc((size_t)PTOT * 128 * 4);
  float* gpool = (float*)alloc((size_t)BB * 1024 * 4);
  float* g512 = (float*)alloc((size_t)BB * 512 * 4);
  float* g256 = (float*)alloc((size_t)BB * 256 * 4);
  float* f3o = (float*)alloc((size_t)BB * 225 * 4);
  float* h512 = (float*)alloc((size_t)PTOT * 512 * 4);
  float* h1024 = (float*)alloc((size_t)PTOT * 1024 * 4);
  float* hfi = (float*)alloc((size_t)PTOT * 1024 * 4);
  float* t512a = (float*)alloc((size_t)PTOT * 512 * 4);
  float* t512b = (float*)alloc((size_t)PTOT * 512 * 4);
  float* t256a = (float*)alloc((size_t)PTOT * 256 * 4);
  float* t256b = (float*)alloc((size_t)PTOT * 256 * 4);
  _Float16* packW = (_Float16*)alloc((size_t)1024 * 1088 * 2);
  float* st = (float*)alloc(2048 * 4);

  // ---- host helpers ----
  auto gemm = [&](const float* A, int lda, const float* Wt, const float* bias, float* Out,
                  long M, int K, int Nc, int ldo, int ocol,
                  const int* eidx, int Knb, int C, int ldx) {
    int Kpad = rup(K, 64), Npad = rup(Nc, 16);
    int tot = Kpad * Npad;
    k_pack<<<cdiv(tot, 256), 256, 0, stream>>>(Wt, packW, K, Nc, Kpad, Npad);
    dim3 grid(cdiv(M, 128), cdiv(Nc, 64));
    if (eidx)
      k_gemm<2><<<grid, 256, 0, stream>>>(A, lda, eidx, Knb, C, ldx, packW, bias, Out,
                                          (int)M, Kpad, K, Nc, Npad, ldo, ocol);
    else
      k_gemm<0><<<grid, 256, 0, stream>>>(A, lda, nullptr, 0, 0, 0, packW, bias, Out,
                                          (int)M, Kpad, K, Nc, Npad, ldo, ocol);
  };
  auto bnstats = [&](const float* Out, long M, int Nc) {
    k_zero<<<cdiv(2 * Nc, 256), 256, 0, stream>>>(st, 2 * Nc);
    k_bnstats<<<512, 256, 0, stream>>>(Out, M, Nc, st);
  };
  auto layer = [&](const float* A, int lda, int pi, float* Out, long M, int K, int Nc, int act) {
    gemm(A, lda, F(pi), F(pi + 1), Out, M, K, Nc, Nc, 0, nullptr, 0, 0, 0);
    bnstats(Out, M, Nc);
    k_bnact<<<cdiv(M * Nc, 256), 256, 0, stream>>>(Out, M, Nc, st, F(pi + 2), F(pi + 3),
                                                   1.f / (float)M, act);
  };
  auto edge = [&](const float* xin, int C, int ldx, const int* eidx, int Knb,
                  int p1, int hid, int p2, int cout, int ocol) {
    long M = (long)PTOT * Knb;
    gemm(xin, 0, F(p1), F(p1 + 1), E1, M, 2 * C, hid, hid, 0, eidx, Knb, C, ldx);
    bnstats(E1, M, hid);
    k_bnact<<<cdiv(M * hid, 256), 256, 0, stream>>>(E1, M, hid, st, F(p1 + 2), F(p1 + 3),
                                                    1.f / (float)M, 1);
    gemm(E1, hid, F(p2), F(p2 + 1), E2, M, hid, cout, cout, 0, nullptr, 0, 0, 0);
    bnstats(E2, M, cout);
    k_bnmaxk<<<cdiv((long)PTOT * cout, 256), 256, 0, stream>>>(
        E2, PTOT, Knb, cout, st, F(p2 + 2), F(p2 + 3), 1.f / (float)M, xcat, 380, ocol);
  };
  auto bld = [&](int* out, int outK, int4 a, int4 b, int4 c, int4 d) {
    k_bldidx<<<cdiv((long)PTOT * outK, 256), 256, 0, stream>>>(fdS, cdS, 544, out, outK, PTOT,
                                                              a, b, c, d);
  };
  int4 NS = make_int4(-1, 0, 0, 0);

  // ---- 0) sort position rows once (shared by all pd/cd selections, max all_k = 540) ----
  k_sortknn<<<PTOT, 256, 0, stream>>>(pos, 3, 3, cdS, 544, 540, NPTS);

  // ---- 1) STN ----
  layer(x, 15, 2, h64, PTOT, 15, 64, 2);
  layer(h64, 64, 6, h128, PTOT, 64, 128, 2);
  layer(h128, 128, 10, E2, PTOT, 128, 1024, 2);
  k_maxpool<<<cdiv(BB * 1024, 256), 256, 0, stream>>>(E2, BB, NPTS, 1024, gpool);
  layer(gpool, 1024, 14, g512, BB, 1024, 512, 2);
  layer(g512, 512, 18, g256, BB, 512, 256, 2);
  gemm(g256, 256, F(22), F(23), f3o, BB, 256, 225, 225, 0, nullptr, 0, 0, 0);
  k_stnapply<<<cdiv(PTOT * 15, 256), 256, 0, stream>>>(x, f3o, xstn, PTOT, NPTS);

  // ---- 2) stage 1: C=15 -> x1 (60ch at xcat col 0) ----
  k_sortknn<<<PTOT, 256, 0, stream>>>(xstn, 15, 15, fdS, 544, 240, NPTS);
  bld(idxb, 44, make_int4(0, 32, 1, 0), make_int4(0, 12, 4, 0), NS, NS);
  bld(idxf, 44, make_int4(0, 32, 7, 0), make_int4(0, 12, 20, 0), NS, NS);
  edge(xstn, 15, 15, idxb, 44, 24, 24, 28, 24, 0);
  edge(xstn, 15, 15, idxf, 44, 32, 36, 36, 36, 24);

  // ---- 3) stage 2: C=60 (xcat cols 0..59) -> x2 (120ch at col 60) ----
  k_sortknn<<<PTOT, 256, 0, stream>>>(xcat, 380, 60, fdS, 544, 360, NPTS);
  bld(idxb, 44, make_int4(0, 16, 1, 0), make_int4(1, 16, 1, 0), make_int4(0, 6, 4, 0),
      make_int4(1, 6, 4, 0));
  bld(idxf, 42, make_int4(0, 18, 20, 0), make_int4(1, 14, 25, 0), make_int4(0, 4, 90, 0),
      make_int4(1, 6, 60, 0));
  edge(xcat, 60, 380, idxb, 44, 40, 48, 44, 48, 60);
  edge(xcat, 60, 380, idxf, 42, 48, 72, 52, 72, 108);

  // ---- 4) stage 3: C=120 (xcat cols 60..179) -> x3 (200ch at col 180) ----
  k_sortknn<<<PTOT, 256, 0, stream>>>(xcat + 60, 380, 120, fdS, 544, 540, NPTS);
  bld(idxb, 44, make_int4(0, 16, 1, 0), make_int4(1, 16, 1, 0), make_int4(0, 6, 4, 0),
      make_int4(1, 6, 4, 0));
  bld(idxf, 42, make_int4(0, 18, 30, 0), make_int4(1, 14, 38, 0), make_int4(0, 4, 135, 0),
      make_int4(1, 6, 90, 0));
  edge(xcat + 60, 120, 380, idxb, 44, 56, 80, 60, 80, 180);
  edge(xcat + 60, 120, 380, idxf, 42, 64, 120, 68, 120, 260);

  // ---- 5) head ----
  layer(xcat, 380, 72, h512, PTOT, 380, 512, 1);    // mlp1
  layer(h512, 512, 76, h1024, PTOT, 512, 1024, 1);  // mlp2
  layer(h1024, 1024, 80, hfi, PTOT, 1024, 1024, 3); // fi (sigmoid)
  k_mul<<<cdiv((long)PTOT * 1024, 256), 256, 0, stream>>>(h1024, hfi, (long)PTOT * 1024);
  // res1
  layer(h1024, 1024, 84, t512a, PTOT, 1024, 512, 1);
  layer(t512a, 512, 88, t512b, PTOT, 512, 512, 1);
  layer(h1024, 1024, 92, h512, PTOT, 1024, 512, 0); // rs (BN, no act)
  k_add<<<cdiv((long)PTOT * 512, 256), 256, 0, stream>>>(t512b, h512, (long)PTOT * 512);
  // res2
  layer(t512b, 512, 96, t256a, PTOT, 512, 256, 1);
  layer(t256a, 256, 100, t256b, PTOT, 256, 256, 1);
  layer(t512b, 512, 104, t256a, PTOT, 512, 256, 0);
  k_add<<<cdiv((long)PTOT * 256, 256), 256, 0, stream>>>(t256b, t256a, (long)PTOT * 256);
  // out: 256 -> 17, bias only
  gemm(t256b, 256, F(108), F(109), (float*)d_out, PTOT, 256, 17, 17, 0, nullptr, 0, 0, 0);
}
